// StackedTransEncoder_85392539779671
// MI455X (gfx1250) — matrix operand order repaired and checked by reading, not compiled, not run on hardware
//
#include <hip/hip_runtime.h>

// ---------------------------------------------------------------------------
// StackedTransEncoder for MI455X (gfx1250): bidirectional GRU with transition
// layers. Recurrent GEMMs use v_wmma_f32_16x16x32_bf16 with pre-packed
// fragment-order operands (weights resident in L2, h re-packed each step).
// GEMM shapes are compile-time so the hot loop is immediate-offset
// global_load_b128 + v_wmma only.
// ---------------------------------------------------------------------------

typedef __bf16 v16bf __attribute__((ext_vector_type(16)));
typedef float  v8f   __attribute__((ext_vector_type(8)));

#define LSEQ 256
#define BATCH 128
#define HID 512
#define EMB 512

// float -> bf16 round-to-nearest-even (bit trick)
__device__ __forceinline__ unsigned short f2bf(float f) {
    unsigned int u = __float_as_uint(f);
    unsigned int r = u + 0x7FFFu + ((u >> 16) & 1u);
    return (unsigned short)(r >> 16);
}

__device__ __forceinline__ float sigm(float x) { return 1.0f / (1.0f + __expf(-x)); }

// A-fragment element mapping (16x32 bf16, M x K), as packed by this file (not the ISA table order):
// lane: m_local = lane&15, hi = lane>>4; elem i:
// k_local = 16*hi + i (contiguous, matches B packing)
__device__ __forceinline__ int a_klocal(int hi, int i) {
    return 16 * hi + i;
}

// ---------------------------------------------------------------------------
// Zero initial hidden state (f32 + packed bf16), both directions.
// ---------------------------------------------------------------------------
__global__ __launch_bounds__(256) void init_h_kernel(
    float* hf0, float* hf1, unsigned short* hp0, unsigned short* hp1) {
    int tid = blockIdx.x * blockDim.x + threadIdx.x;  // 65536
    hf0[tid] = 0.0f; hf1[tid] = 0.0f;
    hp0[tid] = 0;    hp1[tid] = 0;
}

// ---------------------------------------------------------------------------
// Embedding gather -> bf16 packed A-fragments, per timestep.
// ---------------------------------------------------------------------------
__global__ __launch_bounds__(256) void pack_x_kernel(
    const int* __restrict__ xs, const float* __restrict__ emb,
    unsigned short* __restrict__ xpk) {
    int tid = blockIdx.x * blockDim.x + threadIdx.x;  // L*4096
    int t  = tid >> 12;
    int tl = tid & 4095;
    int tile = tl >> 5, lane = tl & 31;
    int mt = tile >> 4, kt = tile & 15;
    int b  = mt * 16 + (lane & 15);
    int hi = lane >> 4;
    int row = xs[t * BATCH + b];
    alignas(16) unsigned short pk[16];
#pragma unroll
    for (int i = 0; i < 16; ++i) {
        int col = kt * 32 + a_klocal(hi, i);
        pk[i] = f2bf(emb[(size_t)row * EMB + col]);
    }
    uint4* d = reinterpret_cast<uint4*>(xpk + ((size_t)t * 4096 + tl) * 16);
    const uint4* s = reinterpret_cast<const uint4*>(pk);
    d[0] = s[0]; d[1] = s[1];
}

// ---------------------------------------------------------------------------
// Pack the stacked lgru weight matrix B' (1024 x 2048) into B-fragments.
// cols [0,1024):  [Uh ; Wx] (z,r fused)
// cols [1024,1536): Uh_n (rows>=512 zero)
// cols [1536,2048): Wx_n (rows<512 zero)
// Tile id = nt*32 + kt; each tile 32x16 bf16 = 512 elems.
// ---------------------------------------------------------------------------
__global__ __launch_bounds__(256) void pack_wl_kernel(
    const float* __restrict__ Uh, const float* __restrict__ Wx,
    unsigned short* __restrict__ dst) {
    int tid = blockIdx.x * blockDim.x + threadIdx.x;  // 131072
    int tile = tid >> 5, lane = tid & 31;
    int nt = tile >> 5, kt = tile & 31;
    int n  = nt * 16 + (lane & 15);
    int hi = lane >> 4;
    alignas(16) unsigned short pk[16];
#pragma unroll
    for (int i = 0; i < 16; ++i) {
        int k = kt * 32 + 16 * hi + i;  // B-layout: lanes 0-15 K 0-15, 16-31 K 16-31
        float v = 0.0f;
        if (n < 1024)      v = (k < 512) ? Uh[(size_t)k * 1536 + n]
                                         : Wx[(size_t)(k - 512) * 1536 + n];
        else if (n < 1536) { if (k < 512)  v = Uh[(size_t)k * 1536 + n]; }
        else               { if (k >= 512) v = Wx[(size_t)(k - 512) * 1536 + (n - 512)]; }
        pk[i] = f2bf(v);
    }
    uint4* d = reinterpret_cast<uint4*>(dst + (size_t)tid * 16);
    const uint4* s = reinterpret_cast<const uint4*>(pk);
    d[0] = s[0]; d[1] = s[1];
}

// ---------------------------------------------------------------------------
// Pack a plain transition weight (512 x 1536) into B-fragments. tile = nt*16+kt.
// ---------------------------------------------------------------------------
__global__ __launch_bounds__(256) void pack_wt_kernel(
    const float* __restrict__ U, unsigned short* __restrict__ dst) {
    int tid = blockIdx.x * blockDim.x + threadIdx.x;  // 49152
    int tile = tid >> 5, lane = tid & 31;
    int nt = tile >> 4, kt = tile & 15;
    int n  = nt * 16 + (lane & 15);
    int hi = lane >> 4;
    alignas(16) unsigned short pk[16];
#pragma unroll
    for (int i = 0; i < 16; ++i) {
        int k = kt * 32 + 16 * hi + i;
        pk[i] = f2bf(U[(size_t)k * 1536 + n]);
    }
    uint4* d = reinterpret_cast<uint4*>(dst + (size_t)tid * 16);
    const uint4* s = reinterpret_cast<const uint4*>(pk);
    d[0] = s[0]; d[1] = s[1];
}

// ---------------------------------------------------------------------------
// Recurrent-step GEMM (templated shape). One wave computes a 64x32 f32 block
// via 8 wmma accumulators. KTTOT = 32 (lgru, A = [h;x]) or 16 (tgru, A = h).
// NT = number of 16-col output tiles (128 or 96). Both directions per launch.
// All hot-loop loads are base + immediate-offset global_load_b128.
// ---------------------------------------------------------------------------
template <int KTTOT, int NT>
__global__ __launch_bounds__(128) void gemm_step_kernel(
    const unsigned short* __restrict__ hA_f, const unsigned short* __restrict__ hA_b,
    const unsigned short* __restrict__ xA_f, const unsigned short* __restrict__ xA_b,
    const unsigned short* __restrict__ B_f,  const unsigned short* __restrict__ B_b,
    float* __restrict__ g_f, float* __restrict__ g_b) {
    constexpr int NBLK = NT >> 1;        // 32-col blocks per direction
    int wave = blockIdx.x * (blockDim.x >> 5) + (threadIdx.x >> 5);
    int lane = threadIdx.x & 31;
    int dir  = (wave >= NT) ? 1 : 0;
    int task = wave - dir * NT;
    int mt0 = (task / NBLK) * 4;   // 0 or 4
    int nt0 = (task % NBLK) * 2;

    const unsigned short* hA = dir ? hA_b : hA_f;
    const unsigned short* Bp = dir ? B_b  : B_f;
    float* g = dir ? g_b : g_f;

    v8f zero = {0.f, 0.f, 0.f, 0.f, 0.f, 0.f, 0.f, 0.f};
    v8f acc[4][2];
#pragma unroll
    for (int mm = 0; mm < 4; ++mm)
#pragma unroll
        for (int nn = 0; nn < 2; ++nn) acc[mm][nn] = zero;

    // Constant per-wave base pointers; inner loops use immediate offsets only.
    const unsigned short* aBase[4];
    const unsigned short* bBase[2];
#pragma unroll
    for (int mm = 0; mm < 4; ++mm)
        aBase[mm] = hA + ((size_t)(mt0 + mm) * 16) * 512 + lane * 16;
#pragma unroll
    for (int nn = 0; nn < 2; ++nn)
        bBase[nn] = Bp + ((size_t)(nt0 + nn) * KTTOT) * 512 + lane * 16;

#pragma unroll 4
    for (int kt = 0; kt < 16; ++kt) {
        v16bf a[4], b[2];
#pragma unroll
        for (int mm = 0; mm < 4; ++mm)
            a[mm] = *reinterpret_cast<const v16bf*>(aBase[mm] + kt * 512);
#pragma unroll
        for (int nn = 0; nn < 2; ++nn)
            b[nn] = *reinterpret_cast<const v16bf*>(bBase[nn] + kt * 512);
#pragma unroll
        for (int mm = 0; mm < 4; ++mm)
#pragma unroll
            for (int nn = 0; nn < 2; ++nn)
                acc[mm][nn] = __builtin_amdgcn_wmma_f32_16x16x32_bf16(
                    false, a[mm], false, b[nn], (short)0, acc[mm][nn], false, false);
    }

    if (KTTOT == 32) {  // x-half of the stacked K (lgru only); B tiles 16..31
        const unsigned short* xA = dir ? xA_b : xA_f;
        const unsigned short* axBase[4];
#pragma unroll
        for (int mm = 0; mm < 4; ++mm)
            axBase[mm] = xA + ((size_t)(mt0 + mm) * 16) * 512 + lane * 16;
#pragma unroll 4
        for (int kt = 0; kt < 16; ++kt) {
            v16bf a[4], b[2];
#pragma unroll
            for (int mm = 0; mm < 4; ++mm)
                a[mm] = *reinterpret_cast<const v16bf*>(axBase[mm] + kt * 512);
#pragma unroll
            for (int nn = 0; nn < 2; ++nn)
                b[nn] = *reinterpret_cast<const v16bf*>(bBase[nn] + (16 + kt) * 512);
#pragma unroll
            for (int mm = 0; mm < 4; ++mm)
#pragma unroll
                for (int nn = 0; nn < 2; ++nn)
                    acc[mm][nn] = __builtin_amdgcn_wmma_f32_16x16x32_bf16(
                        false, a[mm], false, b[nn], (short)0, acc[mm][nn], false, false);
        }
    }

    constexpr int ROWSTRIDE = NT * 16;
    int hi = lane >> 4;
    int nl = lane & 15;
#pragma unroll
    for (int mm = 0; mm < 4; ++mm)
#pragma unroll
        for (int nn = 0; nn < 2; ++nn)
#pragma unroll
            for (int v = 0; v < 8; ++v) {
                int m = (mt0 + mm) * 16 + v + 8 * hi;
                int n = (nt0 + nn) * 16 + nl;
                g[(size_t)m * ROWSTRIDE + n] = acc[mm][nn][v];
            }
}

// ---------------------------------------------------------------------------
// Gating: combines GEMM output chunks + bias, applies GRU update, writes new h
// as f32 (normal layout) AND packed bf16 A-fragments for the next GEMM.
// mode 0: lgru (n = tanh(xn + bn + r*hn));  mode 1: tgru (n = tanh(r*(gn+bn)));
// mode 2: tgru + mask blend against step-start h + write ann output.
// One thread owns exactly one A-fragment lane (16 elements).
// ---------------------------------------------------------------------------
__global__ __launch_bounds__(256) void gate_step_kernel(
    const float* __restrict__ g_f, const float* __restrict__ g_b, int gStride,
    const float* __restrict__ bias_f, const float* __restrict__ bias_b,
    const float* __restrict__ hin_f,  const float* __restrict__ hin_b,
    const float* __restrict__ hst_f,  const float* __restrict__ hst_b,
    float* __restrict__ hout_f, float* __restrict__ hout_b,
    unsigned short* __restrict__ hpk_f, unsigned short* __restrict__ hpk_b,
    int mode,
    const float* __restrict__ mask_f, const float* __restrict__ mask_b,
    float* __restrict__ ann_f, float* __restrict__ ann_b) {
    int tid = blockIdx.x * blockDim.x + threadIdx.x;  // 8192
    int dir = tid >> 12;
    int tl  = tid & 4095;
    const float* g   = dir ? g_b   : g_f;
    const float* bia = dir ? bias_b : bias_f;
    const float* hin = dir ? hin_b : hin_f;
    const float* hst = dir ? hst_b : hst_f;
    float* hout = dir ? hout_b : hout_f;
    unsigned short* hpk = dir ? hpk_b : hpk_f;
    const float* msk = dir ? mask_b : mask_f;
    float* ann = dir ? ann_b : ann_f;

    int tile = tl >> 5, lane = tl & 31;
    int mt = tile >> 4, kt = tile & 15;
    int m  = mt * 16 + (lane & 15);
    int hi = lane >> 4;

    alignas(16) unsigned short pk[16];
#pragma unroll
    for (int i = 0; i < 16; ++i) {
        int k = kt * 32 + a_klocal(hi, i);
        float g0 = g[(size_t)m * gStride + k];
        float g1 = g[(size_t)m * gStride + 512 + k];
        float g2 = g[(size_t)m * gStride + 1024 + k];
        float z = sigm(g0 + bia[k]);
        float r = sigm(g1 + bia[512 + k]);
        float n;
        if (mode == 0) {
            float g3 = g[(size_t)m * gStride + 1536 + k];  // xn chunk
            n = tanhf(g3 + bia[1024 + k] + r * g2);
        } else {
            n = tanhf(r * (g2 + bia[1024 + k]));
        }
        float h  = hin[m * HID + k];
        float hn = (1.0f - z) * h + z * n;
        if (mode == 2) {
            float mk = msk[m];
            float h0 = hst[m * HID + k];
            hn = mk * hn + (1.0f - mk) * h0;
            ann[(size_t)m * 1024 + k] = hn * mk;
        }
        hout[m * HID + k] = hn;
        pk[i] = f2bf(hn);
    }
    uint4* d = reinterpret_cast<uint4*>(hpk + (size_t)tl * 16);
    const uint4* s = reinterpret_cast<const uint4*>(pk);
    d[0] = s[0]; d[1] = s[1];
}

// ---------------------------------------------------------------------------
// Host-side orchestration.
// ---------------------------------------------------------------------------
extern "C" void kernel_launch(void* const* d_in, const int* in_sizes, int n_in,
                              void* d_out, int out_size, void* d_ws, size_t ws_size,
                              hipStream_t stream) {
    (void)in_sizes; (void)n_in; (void)out_size; (void)ws_size;
    const int*   xs      = (const int*)d_in[0];
    const float* xs_mask = (const float*)d_in[1];
    const float* emb     = (const float*)d_in[2];
    const float* f_Wx    = (const float*)d_in[3];
    const float* f_Uh    = (const float*)d_in[4];
    const float* f_b     = (const float*)d_in[5];
    const float* b_Wx    = (const float*)d_in[6];
    const float* b_Uh    = (const float*)d_in[7];
    const float* b_b     = (const float*)d_in[8];
    const float* f_tU    = (const float*)d_in[9];
    const float* f_tb    = (const float*)d_in[10];
    const float* b_tU    = (const float*)d_in[11];
    const float* b_tb    = (const float*)d_in[12];
    float* out = (float*)d_out;

    char* ws = (char*)d_ws;
    size_t off = 0;
    auto take = [&](size_t bytes) -> char* {
        char* p = ws + off;
        off = (off + bytes + 255) & ~(size_t)255;
        return p;
    };

    unsigned short* xpk = (unsigned short*)take((size_t)LSEQ * 65536 * 2);     // 32 MB
    unsigned short* Wl[2];
    Wl[0] = (unsigned short*)take((size_t)32 * 128 * 512 * 2);                 // 4 MB each
    Wl[1] = (unsigned short*)take((size_t)32 * 128 * 512 * 2);
    unsigned short* tUp[2][2];
    for (int d = 0; d < 2; ++d)
        for (int l = 0; l < 2; ++l)
            tUp[d][l] = (unsigned short*)take((size_t)16 * 96 * 512 * 2);      // 1.5 MB each
    float* gbuf[2];
    gbuf[0] = (float*)take((size_t)BATCH * 2048 * 4);                          // 1 MB each
    gbuf[1] = (float*)take((size_t)BATCH * 2048 * 4);
    float *hcur_f[2], *ha_f[2], *hb_f[2];
    unsigned short *hcur_p[2], *ha_p[2], *hb_p[2];
    for (int d = 0; d < 2; ++d) {
        hcur_f[d] = (float*)take((size_t)BATCH * HID * 4);
        ha_f[d]   = (float*)take((size_t)BATCH * HID * 4);
        hb_f[d]   = (float*)take((size_t)BATCH * HID * 4);
        hcur_p[d] = (unsigned short*)take((size_t)BATCH * HID * 2);
        ha_p[d]   = (unsigned short*)take((size_t)BATCH * HID * 2);
        hb_p[d]   = (unsigned short*)take((size_t)BATCH * HID * 2);
    }

    // --- setup: zero h, gather+pack x, pack weights (bf16 fragment order) ---
    init_h_kernel<<<256, 256, 0, stream>>>(hcur_f[0], hcur_f[1], hcur_p[0], hcur_p[1]);
    pack_x_kernel<<<4096, 256, 0, stream>>>(xs, emb, xpk);
    pack_wl_kernel<<<512, 256, 0, stream>>>(f_Uh, f_Wx, Wl[0]);
    pack_wl_kernel<<<512, 256, 0, stream>>>(b_Uh, b_Wx, Wl[1]);
    pack_wt_kernel<<<192, 256, 0, stream>>>(f_tU, tUp[0][0]);
    pack_wt_kernel<<<192, 256, 0, stream>>>(f_tU + (size_t)512 * 1536, tUp[0][1]);
    pack_wt_kernel<<<192, 256, 0, stream>>>(b_tU, tUp[1][0]);
    pack_wt_kernel<<<192, 256, 0, stream>>>(b_tU + (size_t)512 * 1536, tUp[1][1]);

    // --- recurrence: fwd at t=s, bwd at t=L-1-s, both dirs per launch ---
    for (int s = 0; s < LSEQ; ++s) {
        int tf = s, tb = LSEQ - 1 - s;
        const unsigned short* xf = xpk + (size_t)tf * 65536;
        const unsigned short* xb = xpk + (size_t)tb * 65536;

        // lgru: g' = [h;x] @ B'  (K=1024, N=2048): 2 dirs x 128 wave-tasks
        gemm_step_kernel<32, 128><<<64, 128, 0, stream>>>(
            hcur_p[0], hcur_p[1], xf, xb, Wl[0], Wl[1], gbuf[0], gbuf[1]);
        gate_step_kernel<<<32, 256, 0, stream>>>(
            gbuf[0], gbuf[1], 2048, f_b, b_b, hcur_f[0], hcur_f[1],
            (const float*)nullptr, (const float*)nullptr,
            ha_f[0], ha_f[1], ha_p[0], ha_p[1], 0,
            (const float*)nullptr, (const float*)nullptr, (float*)nullptr, (float*)nullptr);

        // tgru layer 0
        gemm_step_kernel<16, 96><<<48, 128, 0, stream>>>(
            ha_p[0], ha_p[1], (const unsigned short*)nullptr, (const unsigned short*)nullptr,
            tUp[0][0], tUp[1][0], gbuf[0], gbuf[1]);
        gate_step_kernel<<<32, 256, 0, stream>>>(
            gbuf[0], gbuf[1], 1536, f_tb, b_tb, ha_f[0], ha_f[1],
            (const float*)nullptr, (const float*)nullptr,
            hb_f[0], hb_f[1], hb_p[0], hb_p[1], 1,
            (const float*)nullptr, (const float*)nullptr, (float*)nullptr, (float*)nullptr);

        // tgru layer 1 + mask blend + ann output
        gemm_step_kernel<16, 96><<<48, 128, 0, stream>>>(
            hb_p[0], hb_p[1], (const unsigned short*)nullptr, (const unsigned short*)nullptr,
            tUp[0][1], tUp[1][1], gbuf[0], gbuf[1]);
        gate_step_kernel<<<32, 256, 0, stream>>>(
            gbuf[0], gbuf[1], 1536, f_tb + 1536, b_tb + 1536, hb_f[0], hb_f[1],
            hcur_f[0], hcur_f[1],
            hcur_f[0], hcur_f[1], hcur_p[0], hcur_p[1], 2,
            xs_mask + (size_t)tf * BATCH, xs_mask + (size_t)tb * BATCH,
            out + (size_t)tf * BATCH * 1024, out + (size_t)tb * BATCH * 1024 + 512);
    }
}
